// HybridQLSTM_65481071406710
// MI455X (gfx1250) — matrix operand order repaired and checked by reading, not compiled, not run on hardware
//
#include <hip/hip_runtime.h>
#include <hip/hip_bf16.h>
#include <math.h>

typedef __attribute__((ext_vector_type(16))) _Float16 v16h;
typedef __attribute__((ext_vector_type(8)))  float    v8f;

#define TSTEPS 256
#define BATCH  1024
#define DIN    128

// =====================================================================
// Kernel 1: WMMA projection.
//   Z[row, n] = sum_d inputs[row, d] * Wg[d, w] + bg[w]
//   row = t*BATCH + b  (M = 262144), n = g*4 + w (N = 16), K = 128.
// One wave32 computes one 16x16 tile via 4x v_wmma_f32_16x16x32_f16.
// =====================================================================
__global__ __launch_bounds__(256) void qlstm_proj_wmma(
    const float* __restrict__ inputs,
    const float* __restrict__ Wf, const float* __restrict__ bf,
    const float* __restrict__ Wi, const float* __restrict__ bi,
    const float* __restrict__ Wu, const float* __restrict__ bu,
    const float* __restrict__ Wo, const float* __restrict__ bo,
    float* __restrict__ Z)
{
    const int lane = threadIdx.x & 31;
    const int wave = blockIdx.x * (blockDim.x >> 5) + (threadIdx.x >> 5);
    const long rowbase = (long)wave * 16;

    // Output column owned by this lane (B/C/D stripe rows across lanes).
    const int n = lane & 15;      // column 0..15
    const int g = n >> 2;         // gate
    const int w = n & 3;          // wire
    const int hi = lane >> 4;     // half-wave select
    const float* Wsel = (g == 0) ? Wf : (g == 1) ? Wi : (g == 2) ? Wu : Wo;
    const float* bsel = (g == 0) ? bf : (g == 1) ? bi : (g == 2) ? bu : bo;

    // A-matrix row owned by this lane.
    const float* arp = inputs + (rowbase + (lane & 15)) * DIN;

    v8f acc = {};
    #pragma unroll
    for (int kk = 0; kk < 4; ++kk) {
        // ---- A fragment: 16x32 f16, lane holds row M=lane&15,
        //      K = kk*32 + hi*8 + {0..7} and + 16 + {0..7}
        const int akoff = kk * 32 + hi * 8;
        const float4 a0 = *(const float4*)(arp + akoff);
        const float4 a1 = *(const float4*)(arp + akoff + 4);
        const float4 a2 = *(const float4*)(arp + akoff + 16);
        const float4 a3 = *(const float4*)(arp + akoff + 20);
        v16h A;
        A[0]  = (_Float16)a0.x; A[1]  = (_Float16)a0.y; A[2]  = (_Float16)a0.z; A[3]  = (_Float16)a0.w;
        A[4]  = (_Float16)a1.x; A[5]  = (_Float16)a1.y; A[6]  = (_Float16)a1.z; A[7]  = (_Float16)a1.w;
        A[8]  = (_Float16)a2.x; A[9]  = (_Float16)a2.y; A[10] = (_Float16)a2.z; A[11] = (_Float16)a2.w;
        A[12] = (_Float16)a3.x; A[13] = (_Float16)a3.y; A[14] = (_Float16)a3.z; A[15] = (_Float16)a3.w;

        // ---- B fragment: 32x16 f16, lane holds column n,
        //      K = kk*32 + hi*8 + {0..7} and + 16 + {0..7}
        const int bkbase = kk * 32 + hi * 8;
        v16h B;
        #pragma unroll
        for (int j = 0; j < 16; ++j)
            B[j] = (_Float16)Wsel[(bkbase + (j & 7) + ((j >> 3) << 4)) * 4 + w];

        acc = __builtin_amdgcn_wmma_f32_16x16x32_f16(
            /*neg_a=*/false, A, /*neg_b=*/false, B,
            /*c_mod=*/(short)0, acc, /*reuse_a=*/false, /*reuse_b=*/false);
    }

    const float bias = bsel[w];
    #pragma unroll
    for (int r = 0; r < 8; ++r) {
        const long m = rowbase + r + hi * 8;   // C/D: VGPR r -> M = r (+8 for hi half)
        Z[m * 16 + n] = acc[r] + bias;
    }
}

// =====================================================================
// Kernel 2: sequential LSTM recurrence.
// 64 lanes per batch element: gate = lib>>4, amplitude lane q = lib&15.
// Statevector lives one-amplitude-per-lane; RX/CNOT via xor-shuffles,
// expectations via a signed Walsh-Hadamard butterfly.
// =====================================================================
__device__ __forceinline__ float shflx(float v, int m) {
    return __shfl_xor(v, m, 32);   // masks < 16 stay inside the 16-lane group
}
__device__ __forceinline__ float sigmoidf_(float x) {
    return 1.0f / (1.0f + __expf(-x));
}

__global__ __launch_bounds__(256) void qlstm_recur(
    const float* __restrict__ Z,
    const float* __restrict__ Wf, const float* __restrict__ thf,
    const float* __restrict__ Wi, const float* __restrict__ thi,
    const float* __restrict__ Wu, const float* __restrict__ thu,
    const float* __restrict__ Wo, const float* __restrict__ tho,
    float* __restrict__ out)   // stacked[T,B,4] ++ hx[B,4] ++ cx[B,4]
{
    __shared__ __align__(16) float s_exps[4][16];  // [batch_in_block][gate*4+w]
    __shared__ __align__(16) float s_hx[4][4];

    const int tid = threadIdx.x;
    const int bib = tid >> 6;      // batch element within block (0..3)
    const int lib = tid & 63;      // lane within batch element
    const int g   = lib >> 4;      // gate: 0=f 1=i 2=u 3=o
    const int q   = lib & 15;      // amplitude index
    const int b   = blockIdx.x * 4 + bib;

    const float* Wsel = (g == 0) ? Wf : (g == 1) ? Wi : (g == 2) ? Wu : Wo;
    const float* tsel = (g == 0) ? thf : (g == 1) ? thi : (g == 2) ? thu : tho;

    // Recurrent weight block (rows 128..131) and trainable-RX cos/sin.
    float wh[4][4];
    #pragma unroll
    for (int k = 0; k < 4; ++k)
        #pragma unroll
        for (int w = 0; w < 4; ++w)
            wh[k][w] = Wsel[(128 + k) * 4 + w];
    float ct[4], st[4];
    #pragma unroll
    for (int w = 0; w < 4; ++w)
        __sincosf(tsel[w] * 0.5f, &st[w], &ct[w]);

    if (lib < 4) s_hx[bib][lib] = 0.0f;
    float cx = 0.0f;               // live only in lanes lib<4 (component = lib)
    __syncthreads();
    float4 hx = *(const float4*)&s_hx[bib][0];

    for (int t = 0; t < TSTEPS; ++t) {
        const long row = (long)t * BATCH + b;
        const float4 z = *(const float4*)(Z + row * 16 + g * 4);
        float ang[4] = {z.x, z.y, z.z, z.w};
        #pragma unroll
        for (int w = 0; w < 4; ++w)
            ang[w] += hx.x * wh[0][w] + hx.y * wh[1][w]
                    + hx.z * wh[2][w] + hx.w * wh[3][w];

        // |0000>
        float ar = (q == 0) ? 1.0f : 0.0f;
        float ai = 0.0f;

        // Encoding RX(ang[w]) on wire w  (wire w <-> index bit 3-w)
        #pragma unroll
        for (int w = 0; w < 4; ++w) {
            float c, s; __sincosf(ang[w] * 0.5f, &s, &c);
            const int m = 8 >> w;
            const float pr = shflx(ar, m), pi = shflx(ai, m);
            const float nar = c * ar + s * pi;   // a' = c*a - i*s*partner
            ai = c * ai - s * pr;
            ar = nar;
        }
        // Trainable RX(theta[w])
        #pragma unroll
        for (int w = 0; w < 4; ++w) {
            const int m = 8 >> w;
            const float pr = shflx(ar, m), pi = shflx(ai, m);
            const float nar = ct[w] * ar + st[w] * pi;
            ai = ct[w] * ai - st[w] * pr;
            ar = nar;
        }
        // CNOT ring (0,1),(1,2),(2,3),(3,0)
        #pragma unroll
        for (int w = 0; w < 4; ++w) {
            const int mc = 8 >> w;
            const int mt = 8 >> ((w + 1) & 3);
            const float pr = shflx(ar, mt), pi = shflx(ai, mt);
            if (q & mc) { ar = pr; ai = pi; }
        }

        // Probabilities + signed WHT: lane (1<<(3-w)) ends with expectation e_w
        float p = ar * ar + ai * ai;
        #pragma unroll
        for (int m = 1; m < 16; m <<= 1) {
            const float pp = shflx(p, m);
            p = (q & m) ? (pp - p) : (p + pp);
        }
        if      (q == 8) s_exps[bib][g * 4 + 0] = p;
        else if (q == 4) s_exps[bib][g * 4 + 1] = p;
        else if (q == 2) s_exps[bib][g * 4 + 2] = p;
        else if (q == 1) s_exps[bib][g * 4 + 3] = p;
        __syncthreads();

        if (lib < 4) {
            const int k = lib;
            const float fv = sigmoidf_(s_exps[bib][0 * 4 + k]);
            const float iv = sigmoidf_(s_exps[bib][1 * 4 + k]);
            const float gv = tanhf    (s_exps[bib][2 * 4 + k]);
            const float ov = sigmoidf_(s_exps[bib][3 * 4 + k]);
            cx = fv * cx + iv * gv;
            const float h = ov * tanhf(cx);
            s_hx[bib][k] = h;
            out[row * 4 + k] = h;                 // stacked[t, b, k]
        }
        __syncthreads();
        hx = *(const float4*)&s_hx[bib][0];
    }

    if (lib < 4) {
        const long base = (long)TSTEPS * BATCH * 4;
        out[base + (long)b * 4 + lib]              = s_hx[bib][lib];  // final hx
        out[base + (long)BATCH * 4 + (long)b * 4 + lib] = cx;        // final cx
    }
}

// =====================================================================
// Launch
// =====================================================================
extern "C" void kernel_launch(void* const* d_in, const int* in_sizes, int n_in,
                              void* d_out, int out_size, void* d_ws, size_t ws_size,
                              hipStream_t stream) {
    (void)in_sizes; (void)n_in; (void)out_size; (void)ws_size;

    const float* inputs = (const float*)d_in[0];
    const float* Wf = (const float*)d_in[1];  const float* bf = (const float*)d_in[2];  const float* thf = (const float*)d_in[3];
    const float* Wi = (const float*)d_in[4];  const float* bi = (const float*)d_in[5];  const float* thi = (const float*)d_in[6];
    const float* Wu = (const float*)d_in[7];  const float* bu = (const float*)d_in[8];  const float* thu = (const float*)d_in[9];
    const float* Wo = (const float*)d_in[10]; const float* bo = (const float*)d_in[11]; const float* tho = (const float*)d_in[12];
    float* out = (float*)d_out;
    float* Z   = (float*)d_ws;   // [T*B, 16] f32 = 16 MB

    // Kernel 1: 262144 rows / 16 per wave = 16384 waves; 8 waves/block.
    qlstm_proj_wmma<<<2048, 256, 0, stream>>>(inputs, Wf, bf, Wi, bi, Wu, bu, Wo, bo, Z);

    // Kernel 2: 1024 batch * 64 lanes = 65536 threads; 4 batch/block.
    qlstm_recur<<<256, 256, 0, stream>>>(Z, Wf, thf, Wi, thi, Wu, thu, Wo, tho, out);
}
